// MoEFFN_85332410237529
// MI455X (gfx1250) — matrix operand order repaired and checked by reading, not compiled, not run on hardware
//
#include <hip/hip_runtime.h>

// ---------------------------------------------------------------------------
// MoE FFN for MI455X (gfx1250, wave32). Strategy:
//   - one memory-bound pass converts x + all GEMM weights fp32 -> bf16 (ws)
//   - all GEMMs use v_wmma_f32_16x16x32_bf16 with pure b128 loads in the loop
//   - sched_group_barrier forces per-k-step "14 loads then 6 WMMAs" issue
//     order so loadcnt waits stagger instead of serializing at 0x0
//   - sparse top-2 routed experts via per-expert token buckets
//   - deterministic: routed contributions land in unique (token,kslot) rows
// ---------------------------------------------------------------------------

typedef __attribute__((ext_vector_type(16))) __bf16 v16bf;
typedef __attribute__((ext_vector_type(8)))  __bf16 v8bf;
typedef __attribute__((ext_vector_type(8)))  float  v8f;

static constexpr int NTOK = 2048;   // B*T
static constexpr int DDIM = 768;
static constexpr int IDIM = 384;
static constexpr int NEXP = 16;
static constexpr int CAP  = 2048;   // per-expert bucket capacity (worst case)

// ---- workspace layout (bytes) ----
static constexpr size_t WS_CNT   = 0;                                // 16 * i32
static constexpr size_t WS_FSUM  = 64;                               // 16 * f32
static constexpr size_t WS_PSUM  = 128;                              // 16 * f32
static constexpr size_t WS_TOK   = 256;                              // 16*2048 u32
static constexpr size_t WS_WGT   = WS_TOK + (size_t)NEXP*CAP*4;      // 16*2048 f32
static constexpr size_t WS_H     = WS_WGT + (size_t)NEXP*CAP*4;      // NTOK*IDIM bf16
static constexpr size_t WS_RH    = WS_H  + (size_t)NTOK*IDIM*2;      // 2*NTOK*IDIM bf16
static constexpr size_t WS_CONTR = WS_RH + (size_t)2*NTOK*IDIM*2;    // 2*NTOK*DDIM f32
static constexpr size_t WS_XB    = WS_CONTR + (size_t)2*NTOK*DDIM*4; // NTOK*DDIM bf16
static constexpr size_t WS_SGB   = WS_XB  + (size_t)NTOK*DDIM*2;     // IDIM*DDIM bf16
static constexpr size_t WS_SUB   = WS_SGB + (size_t)IDIM*DDIM*2;
static constexpr size_t WS_SDB   = WS_SUB + (size_t)IDIM*DDIM*2;     // DDIM*IDIM bf16
static constexpr size_t WS_RGB   = WS_SDB + (size_t)DDIM*IDIM*2;     // E*IDIM*DDIM bf16
static constexpr size_t WS_RUB   = WS_RGB + (size_t)NEXP*IDIM*DDIM*2;
static constexpr size_t WS_RDB   = WS_RUB + (size_t)NEXP*IDIM*DDIM*2;

// sched_group_barrier masks
#define SG_WMMA 0x008
#define SG_VMEM_RD 0x020

__device__ __forceinline__ __bf16 f2bf(float f) {
    unsigned u = __builtin_bit_cast(unsigned, f);
    unsigned r = (u + 0x7FFFu + ((u >> 16) & 1u)) >> 16;      // round-to-nearest-even
    unsigned short h = (unsigned short)r;
    return __builtin_bit_cast(__bf16, h);
}

// A fragment (16x32 bf16): row = lane&15; lanes 0-15 hold K {0..7,16..23},
// lanes 16-31 hold K {8..15,24..31}.  (CDNA5 ISA, 16-bit A layout)
__device__ __forceinline__ v16bf load_a(const __bf16* row, int kc, int lane) {
    int h8 = (lane & 16) ? 8 : 0;
    v8bf r0 = *reinterpret_cast<const v8bf*>(row + kc + h8);
    v8bf r1 = *reinterpret_cast<const v8bf*>(row + kc + h8 + 16);
    v16bf a;
#pragma unroll
    for (int j = 0; j < 8; ++j) { a[j] = r0[j]; a[8 + j] = r1[j]; }
    return a;
}

// B fragment (32x16 bf16): col n = lane&15; lanes 0-15 hold K {0..7,16..23},
// lanes 16-31 hold K {8..15,24..31}.  Weight rows are (out, K) row-major == B^T.
__device__ __forceinline__ v16bf load_b(const __bf16* wrow, int kc, int lane) {
    int h16 = (lane & 16) ? 8 : 0;
    v8bf r0 = *reinterpret_cast<const v8bf*>(wrow + kc + h16);
    v8bf r1 = *reinterpret_cast<const v8bf*>(wrow + kc + h16 + 16);
    v16bf b;
#pragma unroll
    for (int j = 0; j < 8; ++j) { b[j] = r0[j]; b[8 + j] = r1[j]; }
    return b;
}

__device__ __forceinline__ v8f wmma_bf16(v16bf a, v16bf b, v8f c) {
    return __builtin_amdgcn_wmma_f32_16x16x32_bf16(false, a, false, b, (short)0, c, false, false);
}

__device__ __forceinline__ float silu(float g) { return g / (1.0f + __expf(-g)); }

// ---------------------------------------------------------------------------
// K0: fp32 -> bf16 converter, 8 elements/thread (2x b128 load, 1x b128 store).
// ---------------------------------------------------------------------------
__global__ void k_cvt(const float* __restrict__ s, __bf16* __restrict__ d) {
    size_t i = ((size_t)blockIdx.x * blockDim.x + threadIdx.x) * 8;
    float4 q0 = *reinterpret_cast<const float4*>(s + i);
    float4 q1 = *reinterpret_cast<const float4*>(s + i + 4);
    v8bf v;
    v[0] = f2bf(q0.x); v[1] = f2bf(q0.y); v[2] = f2bf(q0.z); v[3] = f2bf(q0.w);
    v[4] = f2bf(q1.x); v[5] = f2bf(q1.y); v[6] = f2bf(q1.z); v[7] = f2bf(q1.w);
    *reinterpret_cast<v8bf*>(d + i) = v;
}

// ---------------------------------------------------------------------------
// K1: gating. one wave per token. fp32 logits -> softmax -> top-2 -> buckets.
// ---------------------------------------------------------------------------
__global__ void k_gate(const float* __restrict__ x, const float* __restrict__ gw,
                       int* __restrict__ cnt, float* __restrict__ fsum,
                       float* __restrict__ psum, unsigned* __restrict__ toklist,
                       float* __restrict__ wgtlist) {
    __shared__ float p_acc[NEXP], f_acc[NEXP];
    if (threadIdx.x < NEXP) { p_acc[threadIdx.x] = 0.f; f_acc[threadIdx.x] = 0.f; }
    __syncthreads();

    const int wave = threadIdx.x >> 5;
    const int lane = threadIdx.x & 31;
    const int tok  = blockIdx.x * 8 + wave;
    const int e    = lane & 15;                  // lanes 16-31 mirror 0-15

    const float* xr = x + (size_t)tok * DDIM;
    const float* wr = gw + (size_t)e * DDIM;
    float acc = 0.f;
#pragma unroll 4
    for (int d = 0; d < DDIM; d += 4) {
        float4 xv = *reinterpret_cast<const float4*>(xr + d);
        float4 wv = *reinterpret_cast<const float4*>(wr + d);
        acc = fmaf(xv.x, wv.x, acc); acc = fmaf(xv.y, wv.y, acc);
        acc = fmaf(xv.z, wv.z, acc); acc = fmaf(xv.w, wv.w, acc);
    }
    float logit = acc * 2.5f;                    // ROUTE_SCALE

    float mx = logit;
#pragma unroll
    for (int off = 8; off >= 1; off >>= 1) mx = fmaxf(mx, __shfl_xor(mx, off));
    float ex = __expf(logit - mx);
    float sm = ex;
#pragma unroll
    for (int off = 8; off >= 1; off >>= 1) sm += __shfl_xor(sm, off);
    float score = ex / sm;

    float s0 = score; int i0 = e;                // top-1, lowest index on ties
#pragma unroll
    for (int off = 8; off >= 1; off >>= 1) {
        float os = __shfl_xor(s0, off); int oi = __shfl_xor(i0, off);
        if (os > s0 || (os == s0 && oi < i0)) { s0 = os; i0 = oi; }
    }
    float s1 = (e == i0) ? -1.0f : score; int i1 = e;   // top-2
#pragma unroll
    for (int off = 8; off >= 1; off >>= 1) {
        float os = __shfl_xor(s1, off); int oi = __shfl_xor(i1, off);
        if (os > s1 || (os == s1 && oi < i1)) { s1 = os; i1 = oi; }
    }
    float wsum = s0 + s1;
    float w0 = s0 / wsum, w1 = s1 / wsum;

    if (lane < 16) atomicAdd(&p_acc[e], score);          // ds_add_f32
    if (lane == 0) {
        atomicAdd(&f_acc[i0], 1.0f);
        atomicAdd(&f_acc[i1], 1.0f);
        int s = atomicAdd(&cnt[i0], 1);
        toklist[i0 * CAP + s] = (unsigned)tok;           // kslot 0
        wgtlist[i0 * CAP + s] = w0;
        int t = atomicAdd(&cnt[i1], 1);
        toklist[i1 * CAP + t] = (unsigned)tok | (1u << 16);
        wgtlist[i1 * CAP + t] = w1;
    }
    __syncthreads();
    if (threadIdx.x < 16) {
        atomicAdd(&fsum[threadIdx.x], f_acc[threadIdx.x]);
        atomicAdd(&psum[threadIdx.x], p_acc[threadIdx.x]);
    }
}

// ---------------------------------------------------------------------------
// K2: shared up/gate. grid NTOK/16; wave covers 48 i-cols (3 tiles) for g & u.
// Per k-step: 14 b128 loads batched, then 6 WMMAs (sched_group_barrier).
// ---------------------------------------------------------------------------
__global__ void k_shared_gu(const __bf16* __restrict__ xb, const __bf16* __restrict__ sgb,
                            const __bf16* __restrict__ sub, __bf16* __restrict__ h) {
    const int wave = threadIdx.x >> 5, lane = threadIdx.x & 31;
    const int m0 = blockIdx.x * 16;
    const int n0 = wave * 48;
    const __bf16* arow = xb + (size_t)(m0 + (lane & 15)) * DDIM;
    const __bf16* grow[3]; const __bf16* urow[3];
#pragma unroll
    for (int t = 0; t < 3; ++t) {
        grow[t] = sgb + (size_t)(n0 + t * 16 + (lane & 15)) * DDIM;
        urow[t] = sub + (size_t)(n0 + t * 16 + (lane & 15)) * DDIM;
    }
    v8f cg[3] = {}, cu[3] = {};
#pragma unroll 2
    for (int kc = 0; kc < DDIM; kc += 32) {
        v16bf a = load_a(arow, kc, lane);
        v16bf bg[3], bu[3];
#pragma unroll
        for (int t = 0; t < 3; ++t) {
            bg[t] = load_b(grow[t], kc, lane);
            bu[t] = load_b(urow[t], kc, lane);
        }
#pragma unroll
        for (int t = 0; t < 3; ++t) {
            cg[t] = wmma_bf16(a, bg[t], cg[t]);
            cu[t] = wmma_bf16(a, bu[t], cu[t]);
        }
        // force issue order: 14-load clause, then 6 WMMAs (staggered waits)
        __builtin_amdgcn_sched_group_barrier(SG_VMEM_RD, 14, 0);
        __builtin_amdgcn_sched_group_barrier(SG_WMMA, 6, 0);
    }
    const int rbase = m0 + ((lane & 16) ? 8 : 0);
#pragma unroll
    for (int t = 0; t < 3; ++t) {
        const int col = n0 + t * 16 + (lane & 15);
#pragma unroll
        for (int r = 0; r < 8; ++r)
            h[(size_t)(rbase + r) * IDIM + col] = f2bf(silu(cg[t][r]) * cu[t][r]);
    }
}

// ---------------------------------------------------------------------------
// K3: shared down. out(base) = h @ sd^T.  grid (NTOK/16, DDIM/256);
// wave covers 32 d-cols (2 tiles).
// ---------------------------------------------------------------------------
__global__ void k_shared_down(const __bf16* __restrict__ h, const __bf16* __restrict__ sdb,
                              float* __restrict__ out) {
    const int wave = threadIdx.x >> 5, lane = threadIdx.x & 31;
    const int m0 = blockIdx.x * 16;
    const int n0 = blockIdx.y * 256 + wave * 32;
    const __bf16* arow = h + (size_t)(m0 + (lane & 15)) * IDIM;
    const __bf16* brow0 = sdb + (size_t)(n0 + (lane & 15)) * IDIM;
    const __bf16* brow1 = sdb + (size_t)(n0 + 16 + (lane & 15)) * IDIM;
    v8f c0 = {}, c1 = {};
#pragma unroll 2
    for (int kc = 0; kc < IDIM; kc += 32) {
        v16bf a  = load_a(arow, kc, lane);
        v16bf b0 = load_b(brow0, kc, lane);
        v16bf b1 = load_b(brow1, kc, lane);
        c0 = wmma_bf16(a, b0, c0);
        c1 = wmma_bf16(a, b1, c1);
        __builtin_amdgcn_sched_group_barrier(SG_VMEM_RD, 6, 0);
        __builtin_amdgcn_sched_group_barrier(SG_WMMA, 2, 0);
    }
    const int rbase = m0 + ((lane & 16) ? 8 : 0);
#pragma unroll
    for (int r = 0; r < 8; ++r) {
        out[(size_t)(rbase + r) * DDIM + n0 + (lane & 15)]      = c0[r];
        out[(size_t)(rbase + r) * DDIM + n0 + 16 + (lane & 15)] = c1[r];
    }
}

// ---------------------------------------------------------------------------
// K4: routed up/gate per expert bucket (gathered rows). grid (NEXP, CAP/16);
// wave covers 48 i-cols (3 tiles) for g & u.
// ---------------------------------------------------------------------------
__global__ void k_routed_gu(const __bf16* __restrict__ xb, const __bf16* __restrict__ rgb,
                            const __bf16* __restrict__ rub, const int* __restrict__ cnt,
                            const unsigned* __restrict__ toklist, __bf16* __restrict__ rh) {
    const int e = blockIdx.x;
    const int ce = cnt[e];
    const int m0 = blockIdx.y * 16;
    if (m0 >= ce) return;
    const int wave = threadIdx.x >> 5, lane = threadIdx.x & 31;
    const int n0 = wave * 48;

    const int aslot = m0 + (lane & 15);
    unsigned aent = (aslot < ce) ? toklist[e * CAP + aslot] : 0u;
    const __bf16* arow = xb + (size_t)(aent & 0xFFFFu) * DDIM;
    const __bf16* grow[3]; const __bf16* urow[3];
#pragma unroll
    for (int t = 0; t < 3; ++t) {
        grow[t] = rgb + ((size_t)e * IDIM + n0 + t * 16 + (lane & 15)) * DDIM;
        urow[t] = rub + ((size_t)e * IDIM + n0 + t * 16 + (lane & 15)) * DDIM;
    }
    v8f cg[3] = {}, cu[3] = {};
#pragma unroll 2
    for (int kc = 0; kc < DDIM; kc += 32) {
        v16bf a = load_a(arow, kc, lane);
        v16bf bg[3], bu[3];
#pragma unroll
        for (int t = 0; t < 3; ++t) {
            bg[t] = load_b(grow[t], kc, lane);
            bu[t] = load_b(urow[t], kc, lane);
        }
#pragma unroll
        for (int t = 0; t < 3; ++t) {
            cg[t] = wmma_bf16(a, bg[t], cg[t]);
            cu[t] = wmma_bf16(a, bu[t], cu[t]);
        }
        __builtin_amdgcn_sched_group_barrier(SG_VMEM_RD, 14, 0);
        __builtin_amdgcn_sched_group_barrier(SG_WMMA, 6, 0);
    }
    const int rbase = m0 + ((lane & 16) ? 8 : 0);
#pragma unroll
    for (int r = 0; r < 8; ++r) {
        int slot = rbase + r;
        if (slot < ce) {
            unsigned ent = toklist[e * CAP + slot];
            size_t row = (size_t)(ent & 0xFFFFu) * 2 + (ent >> 16);
#pragma unroll
            for (int t = 0; t < 3; ++t) {
                int col = n0 + t * 16 + (lane & 15);
                rh[row * IDIM + col] = f2bf(silu(cg[t][r]) * cu[t][r]);
            }
        }
    }
}

// ---------------------------------------------------------------------------
// K5: routed down per expert; w * (rh @ rd^T) -> unique (token,kslot) rows.
// grid (NEXP, CAP/16, DDIM/256); wave covers 32 d-cols (2 tiles).
// ---------------------------------------------------------------------------
__global__ void k_routed_down(const __bf16* __restrict__ rh, const __bf16* __restrict__ rdb,
                              const int* __restrict__ cnt, const unsigned* __restrict__ toklist,
                              const float* __restrict__ wgtlist, float* __restrict__ contrib) {
    const int e = blockIdx.x;
    const int ce = cnt[e];
    const int m0 = blockIdx.y * 16;
    if (m0 >= ce) return;
    const int wave = threadIdx.x >> 5, lane = threadIdx.x & 31;
    const int n0 = blockIdx.z * 256 + wave * 32;

    const int aslot = m0 + (lane & 15);
    unsigned aent = (aslot < ce) ? toklist[e * CAP + aslot] : 0u;
    const __bf16* arow = rh + ((size_t)(aent & 0xFFFFu) * 2 + (aent >> 16)) * IDIM;
    const __bf16* brow0 = rdb + ((size_t)e * DDIM + n0 + (lane & 15)) * IDIM;
    const __bf16* brow1 = rdb + ((size_t)e * DDIM + n0 + 16 + (lane & 15)) * IDIM;

    v8f c0 = {}, c1 = {};
#pragma unroll 2
    for (int kc = 0; kc < IDIM; kc += 32) {
        v16bf a  = load_a(arow, kc, lane);
        v16bf b0 = load_b(brow0, kc, lane);
        v16bf b1 = load_b(brow1, kc, lane);
        c0 = wmma_bf16(a, b0, c0);
        c1 = wmma_bf16(a, b1, c1);
        __builtin_amdgcn_sched_group_barrier(SG_VMEM_RD, 6, 0);
        __builtin_amdgcn_sched_group_barrier(SG_WMMA, 2, 0);
    }
    const int rbase = m0 + ((lane & 16) ? 8 : 0);
#pragma unroll
    for (int r = 0; r < 8; ++r) {
        int slot = rbase + r;
        if (slot < ce) {
            unsigned ent = toklist[e * CAP + slot];
            size_t row = (size_t)(ent & 0xFFFFu) * 2 + (ent >> 16);
            float w = wgtlist[e * CAP + slot];
            contrib[row * DDIM + n0 + (lane & 15)]      = w * c0[r];
            contrib[row * DDIM + n0 + 16 + (lane & 15)] = w * c1[r];
        }
    }
}

// ---------------------------------------------------------------------------
// K6: out += contrib(top1) + contrib(top2), float4 vectorized.
// ---------------------------------------------------------------------------
__global__ void k_combine(float* __restrict__ out, const float* __restrict__ contrib) {
    const int g = blockIdx.x * blockDim.x + threadIdx.x;     // NTOK * (DDIM/4)
    const int tok = g / (DDIM / 4);
    const int j = g - tok * (DDIM / 4);
    float4* o = reinterpret_cast<float4*>(out);
    const float4* c = reinterpret_cast<const float4*>(contrib);
    float4 a  = o[(size_t)tok * (DDIM / 4) + j];
    float4 c0 = c[((size_t)tok * 2 + 0) * (DDIM / 4) + j];
    float4 c1 = c[((size_t)tok * 2 + 1) * (DDIM / 4) + j];
    a.x += c0.x + c1.x; a.y += c0.y + c1.y;
    a.z += c0.z + c1.z; a.w += c0.w + c1.w;
    o[(size_t)tok * (DDIM / 4) + j] = a;
}

// ---------------------------------------------------------------------------
// K7: aux loss = E * sum_e (f_e/N)*(p_e/N)
// ---------------------------------------------------------------------------
__global__ void k_aux(const float* __restrict__ fsum, const float* __restrict__ psum,
                      float* __restrict__ out_aux) {
    const int l = threadIdx.x;
    float v = (l < NEXP) ? (fsum[l] / (float)NTOK) * (psum[l] / (float)NTOK) : 0.f;
#pragma unroll
    for (int off = 16; off >= 1; off >>= 1) v += __shfl_xor(v, off);
    if (l == 0) out_aux[0] = v * (float)NEXP;
}

// ---------------------------------------------------------------------------
extern "C" void kernel_launch(void* const* d_in, const int* in_sizes, int n_in,
                              void* d_out, int out_size, void* d_ws, size_t ws_size,
                              hipStream_t stream) {
    (void)in_sizes; (void)n_in; (void)out_size; (void)ws_size;
    const float* x    = (const float*)d_in[0];
    const float* gw   = (const float*)d_in[1];
    const float* sg   = (const float*)d_in[2];
    const float* su   = (const float*)d_in[3];
    const float* sd   = (const float*)d_in[4];
    const float* rg   = (const float*)d_in[5];
    const float* ru   = (const float*)d_in[6];
    const float* rd   = (const float*)d_in[7];
    float* out = (float*)d_out;

    char* ws = (char*)d_ws;
    int*      cnt     = (int*)(ws + WS_CNT);
    float*    fsum    = (float*)(ws + WS_FSUM);
    float*    psum    = (float*)(ws + WS_PSUM);
    unsigned* toklist = (unsigned*)(ws + WS_TOK);
    float*    wgtlist = (float*)(ws + WS_WGT);
    __bf16*   h       = (__bf16*)(ws + WS_H);
    __bf16*   rh      = (__bf16*)(ws + WS_RH);
    float*    contrib = (float*)(ws + WS_CONTR);
    __bf16*   xb      = (__bf16*)(ws + WS_XB);
    __bf16*   sgb     = (__bf16*)(ws + WS_SGB);
    __bf16*   sub     = (__bf16*)(ws + WS_SUB);
    __bf16*   sdb     = (__bf16*)(ws + WS_SDB);
    __bf16*   rgb     = (__bf16*)(ws + WS_RGB);
    __bf16*   rub     = (__bf16*)(ws + WS_RUB);
    __bf16*   rdb     = (__bf16*)(ws + WS_RDB);

    hipMemsetAsync(ws, 0, 192, stream);  // cnt + fsum + psum

    // one-shot fp32 -> bf16 conversion of all WMMA operands (memory-bound)
    k_cvt<<<(NTOK * DDIM) / 2048, 256, 0, stream>>>(x,  xb);
    k_cvt<<<(IDIM * DDIM) / 2048, 256, 0, stream>>>(sg, sgb);
    k_cvt<<<(IDIM * DDIM) / 2048, 256, 0, stream>>>(su, sub);
    k_cvt<<<(DDIM * IDIM) / 2048, 256, 0, stream>>>(sd, sdb);
    k_cvt<<<(NEXP * IDIM * DDIM) / 2048, 256, 0, stream>>>(rg, rgb);
    k_cvt<<<(NEXP * IDIM * DDIM) / 2048, 256, 0, stream>>>(ru, rub);
    k_cvt<<<(NEXP * DDIM * IDIM) / 2048, 256, 0, stream>>>(rd, rdb);

    k_gate<<<NTOK / 8, 256, 0, stream>>>(x, gw, cnt, fsum, psum, toklist, wgtlist);
    k_shared_gu<<<NTOK / 16, 256, 0, stream>>>(xb, sgb, sub, h);
    k_shared_down<<<dim3(NTOK / 16, DDIM / 256), 256, 0, stream>>>(h, sdb, out);
    k_routed_gu<<<dim3(NEXP, CAP / 16), 256, 0, stream>>>(xb, rgb, rub, cnt, toklist, rh);
    k_routed_down<<<dim3(NEXP, CAP / 16, DDIM / 256), 256, 0, stream>>>(rh, rdb, cnt, toklist, wgtlist, contrib);
    k_combine<<<(NTOK * (DDIM / 4)) / 256, 256, 0, stream>>>(out, contrib);
    k_aux<<<1, 32, 0, stream>>>(fsum, psum, out + (size_t)NTOK * DDIM);
}
